// DeformableTransformerDecoder_85203561218693
// MI455X (gfx1250) — matrix operand order repaired and checked by reading, not compiled, not run on hardware
//
#include <hip/hip_runtime.h>
#include <math.h>
#include <stdint.h>

// ---------------------------------------------------------------------------
// Deformable-DETR transformer decoder for MI455X (gfx1250, wave32, WMMA).
// Dense GEMMs: v_wmma_f32_16x16x32_f16 with async global->LDS f32 staging
// (global_load_async_to_lds_b128 / s_wait_asynccnt), f16 convert on LDS read.
// GEMM templatized on (transB, relu): no uniform branches in the hot loop,
// async transfers use the shared INST_OFFSET field (applies to LDS+global).
// Self-attention: flash-style single-wave kernel, WMMA for QK^T and PV.
// ---------------------------------------------------------------------------

typedef __attribute__((ext_vector_type(16))) _Float16 v16h;
typedef __attribute__((ext_vector_type(4)))  _Float16 v4h;
typedef __attribute__((ext_vector_type(8)))  float    v8f;
typedef __attribute__((ext_vector_type(4)))  float    v4f;

#define WMMA_F16(A, B, C) \
  __builtin_amdgcn_wmma_f32_16x16x32_f16(false, (A), false, (B), (short)0, (C), false, false)

#define D_MODEL 256
#define NQ      900
#define BS      4
#define NH      8
#define HD      32
#define NLV     4
#define NPT     4
#define STOT    13294
#define NLAYERS 6

// CDNA5 async global->LDS copy, 16 bytes per lane, tracked on ASYNCcnt.
// INST_OFFSET is added to both the LDS and the global address (ISA 08 §4.4),
// so one base pair serves a 64-byte run via offset:0/16/32/48.
template<int OFF>
__device__ __forceinline__ void async_b128(unsigned ldsOff, const float* g)
{
  asm volatile("global_load_async_to_lds_b128 %0, %1, off offset:%2"
               :: "v"(ldsOff), "v"((unsigned long long)(uintptr_t)g), "i"(OFF)
               : "memory");
}
__device__ __forceinline__ void async_b128x4(unsigned ldsOff, const float* g)
{
  async_b128<0>(ldsOff, g);
  async_b128<16>(ldsOff, g);
  async_b128<32>(ldsOff, g);
  async_b128<48>(ldsOff, g);
}
__device__ __forceinline__ void wait_async0()
{
  asm volatile("s_wait_asynccnt 0x0" ::: "memory");
}

// ===========================================================================
// Generic GEMM:  C[M,N] = act( A[M,K] * B + bias ),  B is [K,N] (TRANSB=0)
// or [N,K] (TRANSB=1).  Requirements (all call sites satisfy): K%32==0,
// N%64==0.  Ragged M handled by index clamping (garbage rows never stored).
// 64x64 block tile, 4 waves, 16x16x32 WMMA, K step 32.
// ===========================================================================
template<int TRANSB, int RELU>
__global__ __launch_bounds__(128)
void gemm_f16_wmma(const float* __restrict__ A, const float* __restrict__ B,
                   const float* __restrict__ bias, float* __restrict__ C,
                   int M, int N, int K)
{
  constexpr int BROWS = TRANSB ? 64 : 32;
  constexpr int BCOLS = TRANSB ? 36 : 68;
  __shared__ float Af[64][36];        // A tile 64x32 (f32), padded rows
  __shared__ float Bf[BROWS][BCOLS];  // B tile: [n][k] if TRANSB else [k][n]

  const int tid  = threadIdx.x;
  const int lane = tid & 31;
  const int wave = tid >> 5;
  const int m0   = blockIdx.y * 64;
  const int n0   = blockIdx.x * 64;
  const int r    = lane & 15;
  const int hh   = lane >> 4;

  // --- per-thread staging coordinates (branch-free; rows/cols clamped) ---
  const int arow = tid >> 1;
  const int ac0  = (tid & 1) * 16;
  const float* aBase = A + (size_t)min(m0 + arow, M - 1) * K + ac0;
  const unsigned aLds = (unsigned)(uintptr_t)&Af[arow][ac0];

  const float* bBase;
  unsigned bLds;
  size_t bStep;  // global advance (floats) per K-step of 32
  if (TRANSB) {
    const int tn  = tid >> 1;          // n-row of Bf
    const int tk0 = (tid & 1) * 16;
    bBase = B + (size_t)min(n0 + tn, N - 1) * K + tk0;
    bLds  = (unsigned)(uintptr_t)&Bf[tn][tk0];
    bStep = 32;
  } else {
    const int bk  = tid >> 2;          // k-row of Bf
    const int bc0 = (tid & 3) * 16;
    bBase = B + (size_t)bk * N + n0 + bc0;
    bLds  = (unsigned)(uintptr_t)&Bf[bk][bc0];
    bStep = (size_t)32 * N;
  }

  v8f acc[4] = {v8f{}, v8f{}, v8f{}, v8f{}};

  for (int k0 = 0; k0 < K; k0 += 32) {
    async_b128x4(aLds, aBase);         // A tile 64x32 f32
    async_b128x4(bLds, bBase);         // B tile (layout per TRANSB)
    aBase += 32;
    bBase += bStep;
    wait_async0();        // own transfers complete ...
    __syncthreads();      // ... then all waves' transfers visible

    // ---- A fragment: 16x32, lane r = row (wave*16+r), K-halves by hh ----
    v16h af;
#pragma unroll
    for (int j = 0; j < 8; ++j) {
      af[j]     = (_Float16)Af[wave * 16 + r][8 * hh + j];
      af[8 + j] = (_Float16)Af[wave * 16 + r][16 + 8 * hh + j];
    }
    // ---- 4 N-subtiles: B fragment lane r = N-col, K-halves by hh (same as A) ----
#pragma unroll
    for (int t = 0; t < 4; ++t) {
      v16h bf;
#pragma unroll
      for (int j = 0; j < 16; ++j) {
        const int kk = (j < 8) ? 8 * hh + j : 8 + 8 * hh + j;
        bf[j] = (_Float16)(TRANSB ? Bf[(t * 16 + r) % BROWS][kk % BCOLS]
                                  : Bf[kk % BROWS][(t * 16 + r) % BCOLS]);
      }
      acc[t] = WMMA_F16(af, bf, acc[t]);
    }
    __syncthreads();
  }

  // ---- epilogue: C/D layout  M = i + 8*hh (per wave tile),  N = lane&15 ----
#pragma unroll
  for (int t = 0; t < 4; ++t) {
#pragma unroll
    for (int i = 0; i < 8; ++i) {
      int m = m0 + wave * 16 + 8 * hh + i;
      int n = n0 + t * 16 + r;
      if (m < M && n < N) {
        float v = acc[t][i] + bias[n];
        if (RELU) v = fmaxf(v, 0.f);
        C[(size_t)m * N + n] = v;
      }
    }
  }
}

// ===========================================================================
// Flash-style self-attention: one wave handles 16 queries of one (b,h).
// Layout of q/k/v buffers: [(q*BS+b)*256 + h*32 + d].
// ===========================================================================
__global__ __launch_bounds__(32)
void k_attn(const float* __restrict__ qp, const float* __restrict__ kp,
            const float* __restrict__ vp, const float* __restrict__ mask,
            float* __restrict__ out)
{
  const int bh   = blockIdx.y;
  const int b    = bh >> 3;
  const int h    = bh & 7;
  const int q0   = blockIdx.x * 16;
  const int lane = threadIdx.x;
  const int r    = lane & 15;
  const int hh   = lane >> 4;

  __shared__ _Float16 Ks[32][32];
  __shared__ _Float16 Vs[32][32];
  __shared__ float    Ps[16][32];
  __shared__ _Float16 Pp[16][32];
  __shared__ float rowM[16], rowL[16], rowA[16];

  const float scale = 0.17677669529663687f;  // 1/sqrt(32)
  const int  qrow = q0 + r;
  const bool qv   = qrow < NQ;
  const float* qb = qp + ((size_t)min(qrow, NQ - 1) * BS + b) * D_MODEL + h * HD;

  // Q fragment: unconditional 128-bit loads from clamped row, 0/1 validity mul
  const float qm = qv ? scale : 0.f;
  v16h aq;
  {
    v4f qa = *(const v4f*)(qb + 8 * hh);
    v4f qbv = *(const v4f*)(qb + 8 * hh + 4);
    v4f qc = *(const v4f*)(qb + 16 + 8 * hh);
    v4f qd = *(const v4f*)(qb + 16 + 8 * hh + 4);
#pragma unroll
    for (int e = 0; e < 4; ++e) {
      aq[e]      = (_Float16)(qa[e] * qm);
      aq[4 + e]  = (_Float16)(qbv[e] * qm);
      aq[8 + e]  = (_Float16)(qc[e] * qm);
      aq[12 + e] = (_Float16)(qd[e] * qm);
    }
  }
  if (lane < 16) { rowM[lane] = -1e30f; rowL[lane] = 0.f; rowA[lane] = 1.f; }
  v8f o0 = {}, o1 = {};

  for (int kc = 0; kc < NQ; kc += 32) {
    const int  krow = kc + lane;
    const float km  = (krow < NQ) ? 1.f : 0.f;
    const float* kb = kp + ((size_t)min(krow, NQ - 1) * BS + b) * D_MODEL + h * HD;
    const float* vb = vp + ((size_t)min(krow, NQ - 1) * BS + b) * D_MODEL + h * HD;
#pragma unroll
    for (int d = 0; d < 32; d += 4) {
      v4f kf = *(const v4f*)(kb + d);
      v4f vf = *(const v4f*)(vb + d);
      v4h kh, vh;
#pragma unroll
      for (int e = 0; e < 4; ++e) {
        kh[e] = (_Float16)(kf[e] * km);
        vh[e] = (_Float16)(vf[e] * km);
      }
      *(v4h*)&Ks[lane][d] = kh;
      *(v4h*)&Vs[lane][d] = vh;
    }
    __syncthreads();

    // scores = Q * K^T : B[k=dim][n=key] -> lane%16 = key, dims by hh (same as A)
    v16h bk0, bk1;
#pragma unroll
    for (int j = 0; j < 16; ++j) { const int kk = (j < 8) ? 8 * hh + j : 8 + 8 * hh + j; bk0[j] = Ks[r][kk]; bk1[j] = Ks[16 + r][kk]; }
    v8f s0 = {}, s1 = {};
    s0 = WMMA_F16(aq, bk0, s0);
    s1 = WMMA_F16(aq, bk1, s1);

#pragma unroll
    for (int i = 0; i < 8; ++i) {
      int m   = 8 * hh + i;
      int gq  = q0 + m;
      int gk0 = kc + r;
      int gk1 = kc + 16 + r;
      float m0v = mask[(size_t)min(gq, NQ - 1) * NQ + min(gk0, NQ - 1)];
      float m1v = mask[(size_t)min(gq, NQ - 1) * NQ + min(gk1, NQ - 1)];
      m0v = (gq < NQ && gk0 < NQ) ? m0v : -1e30f;
      m1v = (gq < NQ && gk1 < NQ) ? m1v : -1e30f;
      Ps[m][r]      = s0[i] + m0v;
      Ps[m][16 + r] = s1[i] + m1v;
    }
    __syncthreads();

    if (lane < 16) {  // online softmax update, one lane per query row
      float mx = rowM[lane];
#pragma unroll
      for (int c = 0; c < 32; ++c) mx = fmaxf(mx, Ps[lane][c]);
      float alpha = expf(rowM[lane] - mx);
      float sum = 0.f;
#pragma unroll
      for (int c = 0; c < 32; ++c) {
        float p = expf(Ps[lane][c] - mx);
        Pp[lane][c] = (_Float16)p;
        sum += p;
      }
      rowL[lane] = rowL[lane] * alpha + sum;
      rowM[lane] = mx;
      rowA[lane] = alpha;
    }
    __syncthreads();

#pragma unroll
    for (int i = 0; i < 8; ++i) {
      float al = rowA[8 * hh + i];
      o0[i] *= al; o1[i] *= al;
    }
    // PV : A = P (16x32), B = V chunk (32 keys x 32 dims)
    v16h ap;
#pragma unroll
    for (int j = 0; j < 8; ++j) {
      ap[j]     = Pp[r][8 * hh + j];
      ap[8 + j] = Pp[r][16 + 8 * hh + j];
    }
    v16h bv0, bv1;
#pragma unroll
    for (int j = 0; j < 16; ++j) { const int kk = (j < 8) ? 8 * hh + j : 8 + 8 * hh + j; bv0[j] = Vs[kk][r]; bv1[j] = Vs[kk][16 + r]; }
    o0 = WMMA_F16(ap, bv0, o0);
    o1 = WMMA_F16(ap, bv1, o1);
    __syncthreads();
  }

#pragma unroll
  for (int i = 0; i < 8; ++i) {
    int m  = 8 * hh + i;
    int gq = q0 + m;
    if (gq < NQ) {
      float inv = 1.f / rowL[m];
      float* ob = out + ((size_t)gq * BS + b) * D_MODEL + h * HD;
      ob[r]      = o0[i] * inv;
      ob[16 + r] = o1[i] * inv;
    }
  }
}

// ===========================================================================
// Element-wise / normalization helpers
// ===========================================================================
__global__ void k_sigmoid(const float* __restrict__ a, float* __restrict__ o, int n)
{
  int i = blockIdx.x * 256 + threadIdx.x;
  if (i < n) o[i] = 1.f / (1.f + expf(-a[i]));
}

__global__ void k_add(const float* __restrict__ a, const float* __restrict__ b,
                      float* __restrict__ o, int n)
{
  int i = blockIdx.x * 256 + threadIdx.x;
  if (i < n) o[i] = a[i] + b[i];
}

// qse[(q*BS+b)*512 + k]: order (y, x, w, h), each 128-dim sine embedding.
__global__ __launch_bounds__(256)
void k_sine_embed(const float* __restrict__ ref, const float* __restrict__ vr,
                  float* __restrict__ out)
{
  int gidx = blockIdx.x * 256 + threadIdx.x;
  if (gidx >= NQ * BS * 512) return;
  int k   = gidx & 511;
  int row = gidx >> 9;           // q*BS + b
  int b   = row & (BS - 1);
  float vx = vr[(b * NLV + 0) * 2 + 0];
  float vy = vr[(b * NLV + 0) * 2 + 1];
  const float* r4 = ref + (size_t)row * 4;
  int g = k >> 7;                // 0:y 1:x 2:w 3:h
  float comp = (g == 0) ? r4[1] * vy :
               (g == 1) ? r4[0] * vx :
               (g == 2) ? r4[2] * vx : r4[3] * vy;
  int kk = k & 127;
  int j  = kk >> 1;
  float t = powf(10000.f, (float)j * (1.0f / 64.0f));
  float p = comp * 6.283185307179586f / t;
  out[gidx] = (kk & 1) ? cosf(p) : sinf(p);
}

// out = LN(a (+ b)) * g + beta.  mode 1: store transposed into hs[layer].
__global__ __launch_bounds__(256)
void k_ln(const float* __restrict__ a, const float* __restrict__ b,
          const float* __restrict__ g, const float* __restrict__ beta,
          float* __restrict__ out, int mode, int layer)
{
  int row = blockIdx.x, t = threadIdx.x;
  float x = a[(size_t)row * 256 + t];
  if (b) x += b[(size_t)row * 256 + t];
  __shared__ float red[256];
  red[t] = x; __syncthreads();
  for (int s = 128; s > 0; s >>= 1) { if (t < s) red[t] += red[t + s]; __syncthreads(); }
  float mean = red[0] * (1.f / 256.f);
  __syncthreads();
  float d = x - mean;
  red[t] = d * d; __syncthreads();
  for (int s = 128; s > 0; s >>= 1) { if (t < s) red[t] += red[t + s]; __syncthreads(); }
  float var = red[0] * (1.f / 256.f);
  float y = d * rsqrtf(var + 1e-5f) * g[t] + beta[t];
  if (mode == 0) out[(size_t)row * 256 + t] = y;
  else {
    int q = row / BS, bb = row % BS;
    out[(((size_t)layer * BS + bb) * NQ + q) * 256 + t] = y;
  }
}

__global__ void k_mask_value(float* __restrict__ v, const unsigned char* __restrict__ kpm)
{
  int i = blockIdx.x * 256 + threadIdx.x;
  if (i >= STOT * BS * 256) return;
  int row = i >> 8;
  int s = row / BS, b = row % BS;
  if (kpm[(size_t)b * STOT + s]) v[i] = 0.f;
}

// softmax over the 16 (level,point) weights per head; aw row = [h][l][p]
__global__ __launch_bounds__(128)
void k_aw_softmax(float* __restrict__ aw)
{
  int row = blockIdx.x, t = threadIdx.x;
  __shared__ float s[128];
  float v = aw[(size_t)row * 128 + t];
  s[t] = v; __syncthreads();
  int base = (t >> 4) * 16;
  float mx = -1e30f;
  for (int i = 0; i < 16; ++i) mx = fmaxf(mx, s[base + i]);
  float e = expf(v - mx);
  __syncthreads(); s[t] = e; __syncthreads();
  float sum = 0.f;
  for (int i = 0; i < 16; ++i) sum += s[base + i];
  aw[(size_t)row * 128 + t] = e / sum;
}

// deformable bilinear sampling: one wave per (q,b,h), lane = channel.
__global__ __launch_bounds__(32)
void k_deform(const float* __restrict__ vmem, const float* __restrict__ ref,
              const float* __restrict__ vr, const float* __restrict__ off,
              const float* __restrict__ aw, const int* __restrict__ shapes,
              float* __restrict__ out)
{
  int gid  = blockIdx.x;       // (q*BS+b)*NH + h
  int h    = gid & 7;
  int rb   = gid >> 3;         // q*BS + b
  int b    = rb & (BS - 1);
  int lane = threadIdx.x;
  const float* r4 = ref + (size_t)rb * 4;
  float acc = 0.f;
  int start = 0;
  for (int l = 0; l < NLV; ++l) {
    int H = shapes[l * 2 + 0], W = shapes[l * 2 + 1];
    float vx = vr[(b * NLV + l) * 2 + 0];
    float vy = vr[(b * NLV + l) * 2 + 1];
    float cx = r4[0] * vx, cy = r4[1] * vy;
    float ww = r4[2] * vx, wh = r4[3] * vy;
    for (int p = 0; p < NPT; ++p) {
      int ob = rb * 256 + ((h * NLV + l) * NPT + p) * 2;
      float lx = cx + off[ob + 0] * 0.125f * ww;   // off/NPT * wh * 0.5
      float ly = cy + off[ob + 1] * 0.125f * wh;
      float wgt = aw[(size_t)rb * 128 + (h * 16 + l * 4 + p)];
      float x = lx * W - 0.5f;
      float y = ly * H - 0.5f;
      float xf = floorf(x), yf = floorf(y);
      int x0 = (int)xf, y0 = (int)yf;
      float fx = x - xf, fy = y - yf;
      float w00 = (1.f - fx) * (1.f - fy), w10 = fx * (1.f - fy);
      float w01 = (1.f - fx) * fy,         w11 = fx * fy;
#pragma unroll
      for (int cy2 = 0; cy2 < 2; ++cy2)
#pragma unroll
        for (int cx2 = 0; cx2 < 2; ++cx2) {
          int xi = x0 + cx2, yi = y0 + cy2;
          float cw = cx2 ? (cy2 ? w11 : w10) : (cy2 ? w01 : w00);
          bool valid = (xi >= 0 && xi < W && yi >= 0 && yi < H);
          int xc = min(max(xi, 0), W - 1), yc = min(max(yi, 0), H - 1);
          float gval = vmem[((size_t)(start + yc * W + xc) * BS + b) * 256 + h * HD + lane];
          acc += wgt * cw * (valid ? gval : 0.f);
        }
    }
    start += H * W;
  }
  out[(size_t)rb * 256 + h * HD + lane] = acc;
}

// ref [nq,bs,4] -> out [1,bs,nq,4]
__global__ void k_ref_out(const float* __restrict__ ref, float* __restrict__ outr)
{
  int i = blockIdx.x * 256 + threadIdx.x;
  if (i >= NQ * BS * 4) return;
  int c = i & 3, row = i >> 2;
  int b = row % BS, q = row / BS;
  outr[((size_t)b * NQ + q) * 4 + c] = ref[i];
}

// ===========================================================================
// Host orchestration
// ===========================================================================
struct LayerP {
  const float *aw_b, *aw_w, *co_b, *co_w, *f1_b, *f1_w, *f2_b, *f2_w,
              *n1_b, *n1_w, *n2_b, *n2_w, *n3_b, *n3_w, *off_b, *off_w,
              *sa_in_b, *sa_in_w, *sa_out_b, *sa_out_w, *val_b, *val_w;
};

static inline void gemm(const float* A, const float* B, const float* bias, float* C,
                        int M, int N, int K, int transB, int relu, hipStream_t s)
{
  dim3 g((N + 63) / 64, (M + 63) / 64);
  if (transB) {
    if (relu) gemm_f16_wmma<1, 1><<<g, 128, 0, s>>>(A, B, bias, C, M, N, K);
    else      gemm_f16_wmma<1, 0><<<g, 128, 0, s>>>(A, B, bias, C, M, N, K);
  } else {
    if (relu) gemm_f16_wmma<0, 1><<<g, 128, 0, s>>>(A, B, bias, C, M, N, K);
    else      gemm_f16_wmma<0, 0><<<g, 128, 0, s>>>(A, B, bias, C, M, N, K);
  }
}

extern "C" void kernel_launch(void* const* d_in, const int* in_sizes, int n_in,
                              void* d_out, int out_size, void* d_ws, size_t ws_size,
                              hipStream_t stream)
{
  (void)in_sizes; (void)out_size; (void)ws_size;
  if (n_in < 145) return;  // expected jax pytree flatten: 5 + 6*22 + 2 + 4 + 2

  const float* tgt    = (const float*)d_in[0];
  const float* memory = (const float*)d_in[1];
  const float* refp   = (const float*)d_in[2];
  const float* vr     = (const float*)d_in[3];
  const float* tmask  = (const float*)d_in[4];

  // params: dicts flattened with sorted keys; 'layers' list in order, each layer
  // dict alphabetical, then norm_b, norm_w, then ref_head {b1,b2,w1,w2}.
  int idx = 5;
  auto F = [&](void) { return (const float*)d_in[idx++]; };
  LayerP L[NLAYERS];
  for (int l = 0; l < NLAYERS; ++l) {
    L[l].aw_b = F();     L[l].aw_w = F();     L[l].co_b = F();     L[l].co_w = F();
    L[l].f1_b = F();     L[l].f1_w = F();     L[l].f2_b = F();     L[l].f2_w = F();
    L[l].n1_b = F();     L[l].n1_w = F();     L[l].n2_b = F();     L[l].n2_w = F();
    L[l].n3_b = F();     L[l].n3_w = F();     L[l].off_b = F();    L[l].off_w = F();
    L[l].sa_in_b = F();  L[l].sa_in_w = F();  L[l].sa_out_b = F(); L[l].sa_out_w = F();
    L[l].val_b = F();    L[l].val_w = F();
  }
  const float* norm_b = F();
  const float* norm_w = F();
  const float* rh_b1  = F();
  const float* rh_b2  = F();
  const float* rh_w1  = F();
  const float* rh_w2  = F();
  const int* shapes        = (const int*)d_in[idx++];
  const unsigned char* kpm = (const unsigned char*)d_in[idx++];

  // --- workspace bump allocator ---
  char* wp = (char*)d_ws;
  auto wsAlloc = [&](size_t nFloats) -> float* {
    float* p = (float*)wp;
    wp += ((nFloats * sizeof(float) + 255) & ~(size_t)255);
    return p;
  };
  const int M3600 = NQ * BS;
  float* refS  = wsAlloc((size_t)M3600 * 4);
  float* qse   = wsAlloc((size_t)M3600 * 512);
  float* qh    = wsAlloc((size_t)M3600 * 256);
  float* qpos  = wsAlloc((size_t)M3600 * 256);
  float* qk    = wsAlloc((size_t)M3600 * 256);
  float* qproj = wsAlloc((size_t)M3600 * 256);
  float* kproj = wsAlloc((size_t)M3600 * 256);
  float* vproj = wsAlloc((size_t)M3600 * 256);
  float* attno = wsAlloc((size_t)M3600 * 256);
  float* sa    = wsAlloc((size_t)M3600 * 256);
  float* xbuf  = wsAlloc((size_t)M3600 * 256);
  float* cain  = wsAlloc((size_t)M3600 * 256);
  float* vmem  = wsAlloc((size_t)STOT * BS * 256);
  float* offb  = wsAlloc((size_t)M3600 * 256);
  float* awb   = wsAlloc((size_t)M3600 * 128);
  float* samp  = wsAlloc((size_t)M3600 * 256);
  float* ca    = wsAlloc((size_t)M3600 * 256);
  float* x2    = wsAlloc((size_t)M3600 * 256);
  float* ffh   = wsAlloc((size_t)M3600 * 1024);
  float* ff    = wsAlloc((size_t)M3600 * 256);
  float* outx  = wsAlloc((size_t)M3600 * 256);

  float* hs_out  = (float*)d_out;
  float* ref_out = hs_out + (size_t)NLAYERS * BS * NQ * 256;

  // --- pre-loop: ref sigmoid, sine embed, query-pos MLP, ref output ---
  k_sigmoid<<<(M3600 * 4 + 255) / 256, 256, 0, stream>>>(refp, refS, M3600 * 4);
  k_sine_embed<<<(M3600 * 512 + 255) / 256, 256, 0, stream>>>(refS, vr, qse);
  gemm(qse, rh_w1, rh_b1, qh,   M3600, 256, 512, 0, 1, stream);
  gemm(qh,  rh_w2, rh_b2, qpos, M3600, 256, 256, 0, 0, stream);
  k_ref_out<<<(M3600 * 4 + 255) / 256, 256, 0, stream>>>(refS, ref_out);

  // --- decoder layers ---
  for (int l = 0; l < NLAYERS; ++l) {
    const LayerP& P = L[l];
    const float* cur = (l == 0) ? tgt : outx;

    // self-attention: q = k = cur + qpos, v = cur
    k_add<<<(M3600 * 256 + 255) / 256, 256, 0, stream>>>(cur, qpos, qk, M3600 * 256);
    gemm(qk,  P.sa_in_w,             P.sa_in_b,       qproj, M3600, 256, 256, 1, 0, stream);
    gemm(qk,  P.sa_in_w + 256 * 256, P.sa_in_b + 256, kproj, M3600, 256, 256, 1, 0, stream);
    gemm(cur, P.sa_in_w + 512 * 256, P.sa_in_b + 512, vproj, M3600, 256, 256, 1, 0, stream);
    k_attn<<<dim3((NQ + 15) / 16, BS * NH), 32, 0, stream>>>(qproj, kproj, vproj, tmask, attno);
    gemm(attno, P.sa_out_w, P.sa_out_b, sa, M3600, 256, 256, 0, 0, stream);
    k_ln<<<M3600, 256, 0, stream>>>(cur, sa, P.n2_w, P.n2_b, xbuf, 0, 0);

    // deformable cross-attention
    k_add<<<(M3600 * 256 + 255) / 256, 256, 0, stream>>>(xbuf, qpos, cain, M3600 * 256);
    gemm(memory, P.val_w, P.val_b, vmem, STOT * BS, 256, 256, 0, 0, stream);
    k_mask_value<<<(STOT * BS * 256 + 255) / 256, 256, 0, stream>>>(vmem, kpm);
    gemm(cain, P.off_w, P.off_b, offb, M3600, 256, 256, 0, 0, stream);
    gemm(cain, P.aw_w,  P.aw_b,  awb,  M3600, 128, 256, 0, 0, stream);
    k_aw_softmax<<<M3600, 128, 0, stream>>>(awb);
    k_deform<<<M3600 * NH, 32, 0, stream>>>(vmem, refS, vr, offb, awb, shapes, samp);
    gemm(samp, P.co_w, P.co_b, ca, M3600, 256, 256, 0, 0, stream);
    k_ln<<<M3600, 256, 0, stream>>>(xbuf, ca, P.n1_w, P.n1_b, x2, 0, 0);

    // FFN
    gemm(x2,  P.f1_w, P.f1_b, ffh, M3600, 1024, 256,  0, 1, stream);
    gemm(ffh, P.f2_w, P.f2_b, ff,  M3600, 256,  1024, 0, 0, stream);
    k_ln<<<M3600, 256, 0, stream>>>(x2, ff, P.n3_w, P.n3_b, outx, 0, 0);

    // per-layer normalized output -> hs[l] ([bs, nq, d] layout)
    k_ln<<<M3600, 256, 0, stream>>>(outx, nullptr, norm_w, norm_b, hs_out, 1, l);
  }
}
